// GRN_3564822856102
// MI455X (gfx1250) — matrix operand order repaired and checked by reading, not compiled, not run on hardware
//
#include <hip/hip_runtime.h>
#include <hip/hip_bf16.h>
#include <math.h>

#define DDIM 160
#define KSLABS 5                         // 160 / 32
#define ETILES 10                        // 160 / 16
#define TOKENS (256 * 512)
#define WG_WAVES 8
#define WG_THREADS (WG_WAVES * 32)
#define TILE_M 16
#define NTILES (TOKENS / TILE_M)         // 8192
#define NGROUPS (NTILES / WG_WAVES)      // 1024
#define WELEMS (DDIM * DDIM)             // 25600
#define WALL_BYTES (5 * WELEMS * 2)      // 256000
#define STG_BYTES (TILE_M * DDIM * 2)    // 5120 per wave
#define LDS_TOTAL (WALL_BYTES + WG_WAVES * STG_BYTES)  // 296960 <= 320KB

typedef __attribute__((ext_vector_type(16))) __bf16 v16bf;
typedef __attribute__((ext_vector_type(4)))  __bf16 v4bf;
typedef __attribute__((ext_vector_type(8)))  float  v8f;

union Frag16 { v16bf v; uint4 q[2]; };

__device__ __forceinline__ v8f wmma_bf16(v16bf a, v16bf b, v8f c) {
  // D = A(16x32 bf16) * B(32x16 bf16) + C(16x16 f32)
  return __builtin_amdgcn_wmma_f32_16x16x32_bf16(
      /*neg_a=*/false, a, /*neg_b=*/false, b,
      /*c_mod=*/(short)0, c, /*reuse_a=*/false, /*reuse_b=*/false);
}

// A-matrix 16x32 bf16 fragment from row-major [16][160] bf16 LDS tile.
// lane 0-15: row M=lane, K = kb..kb+15 (kb = 0)
// lane 16-31: same rows, kb = 16.
__device__ __forceinline__ v16bf load_afrag(const __bf16* s, int k, int lane) {
  const char* p = (const char*)s + (lane & 15) * (DDIM * 2) + k * 64 + (lane >> 4) * 32;
  Frag16 f;
  f.q[0] = *(const uint4*)(p);        // K = k*32 + kb + 0..7
  f.q[1] = *(const uint4*)(p + 16);   // K = k*32 + kb + 8..15
  return f.v;
}

// B-matrix 32x16 bf16 fragment (B[k][n] = W[n][k]) from row-major [160][160] W in LDS.
// lane holds column n = lane&15 (weight row e*16+n), K = (lane>>4)*16 + 0..15: 32 contiguous bytes.
__device__ __forceinline__ v16bf load_bfrag(const __bf16* w, int e, int k, int lane) {
  const char* p = (const char*)w + (e * 16 + (lane & 15)) * (DDIM * 2)
                + k * 64 + (lane >> 4) * 32;
  Frag16 f;
  f.q[0] = *(const uint4*)(p);
  f.q[1] = *(const uint4*)(p + 16);
  return f.v;
}

// Convert one 16x160 f32 tile (rows contiguous in global) to bf16 row-major LDS.
__device__ __forceinline__ void stage_tile(__bf16* dst, const float* src, int lane) {
  const float4* s4 = (const float4*)src;
  for (int i = lane; i < (TILE_M * DDIM / 4); i += 32) {
    float4 v = s4[i];
    v4bf p;
    p[0] = (__bf16)v.x; p[1] = (__bf16)v.y; p[2] = (__bf16)v.z; p[3] = (__bf16)v.w;
    *(v4bf*)(dst + 4 * i) = p;
  }
}

// Scatter a 16x16 f32 C fragment back to the row-major bf16 staging tile.
__device__ __forceinline__ void store_cstage(__bf16* stg, v8f a, int e, int lane) {
  int col = e * 16 + (lane & 15);
  int mb = (lane >> 4) * 8;
#pragma unroll
  for (int r = 0; r < 8; ++r) stg[(mb + r) * DDIM + col] = (__bf16)a[r];
}

__global__ __launch_bounds__(WG_THREADS) void grn_fused(
    const float* __restrict__ x, const float* __restrict__ c,
    const float* __restrict__ W2, const float* __restrict__ b2,
    const float* __restrict__ W3,
    const float* __restrict__ W1, const float* __restrict__ b1,
    const float* __restrict__ W4, const float* __restrict__ b4,
    const float* __restrict__ W5, const float* __restrict__ b5,
    const float* __restrict__ gamma, const float* __restrict__ beta,
    float* __restrict__ out)
{
  extern __shared__ char smem[];
  __bf16* wl = (__bf16*)smem;
  const int tid  = threadIdx.x;
  const int lane = tid & 31;
  const int wv   = tid >> 5;

  // Cooperatively convert all 5 weight matrices f32 -> bf16 into LDS (stays resident).
  {
    const float* src[5] = {W2, W3, W1, W4, W5};
#pragma unroll
    for (int m = 0; m < 5; ++m) {
      const float* s = src[m];
      __bf16* d = wl + m * WELEMS;
      for (int i = tid; i < WELEMS; i += WG_THREADS) d[i] = (__bf16)s[i];
    }
  }
  __syncthreads();

  const __bf16* W2l = wl + 0 * WELEMS;
  const __bf16* W3l = wl + 1 * WELEMS;
  const __bf16* W1l = wl + 2 * WELEMS;
  const __bf16* W4l = wl + 3 * WELEMS;
  const __bf16* W5l = wl + 4 * WELEMS;
  __bf16* stg = (__bf16*)(smem + WALL_BYTES + wv * STG_BYTES);
  const int ncol = lane & 15;
  const int mb   = (lane >> 4) * 8;

  for (int g = blockIdx.x; g < NGROUPS; g += gridDim.x) {
    const int tok0 = (g * WG_WAVES + wv) * TILE_M;
    const float* xt = x + (size_t)tok0 * DDIM;
    const float* ct = c + (size_t)tok0 * DDIM;

    // ---------------- eta2 = ELU(x @ W2^T + b2 + c @ W3^T) ----------------
    stage_tile(stg, xt, lane);
    __syncthreads();
    v16bf ax[KSLABS];
#pragma unroll
    for (int k = 0; k < KSLABS; ++k) ax[k] = load_afrag(stg, k, lane);
    __syncthreads();
    stage_tile(stg, ct, lane);
    __syncthreads();
    v16bf ac[KSLABS];
#pragma unroll
    for (int k = 0; k < KSLABS; ++k) ac[k] = load_afrag(stg, k, lane);
    __syncthreads();

#pragma unroll
    for (int e = 0; e < ETILES; ++e) {
      v8f acc = {};
#pragma unroll
      for (int k = 0; k < KSLABS; ++k) acc = wmma_bf16(ax[k], load_bfrag(W2l, e, k, lane), acc);
#pragma unroll
      for (int k = 0; k < KSLABS; ++k) acc = wmma_bf16(ac[k], load_bfrag(W3l, e, k, lane), acc);
      float bb = b2[e * 16 + ncol];
#pragma unroll
      for (int r = 0; r < 8; ++r) {
        float v = acc[r] + bb;
        acc[r] = v > 0.0f ? v : (__expf(v) - 1.0f);   // ELU
      }
      store_cstage(stg, acc, e, lane);
    }
    __syncthreads();

    // ---------------- eta1 = eta2 @ W1^T + b1 ----------------
    v16bf ae[KSLABS];
#pragma unroll
    for (int k = 0; k < KSLABS; ++k) ae[k] = load_afrag(stg, k, lane);
    __syncthreads();
#pragma unroll
    for (int e = 0; e < ETILES; ++e) {
      v8f acc = {};
#pragma unroll
      for (int k = 0; k < KSLABS; ++k) acc = wmma_bf16(ae[k], load_bfrag(W1l, e, k, lane), acc);
      float bb = b1[e * 16 + ncol];
#pragma unroll
      for (int r = 0; r < 8; ++r) acc[r] += bb;
      store_cstage(stg, acc, e, lane);
    }
    __syncthreads();

    // ---------------- GLU + residual + LayerNorm ----------------
    v16bf ah[KSLABS];
#pragma unroll
    for (int k = 0; k < KSLABS; ++k) ah[k] = load_afrag(stg, k, lane);
    __syncthreads();

    v8f yv[ETILES];
    float rs[8], rq[8];
#pragma unroll
    for (int r = 0; r < 8; ++r) { rs[r] = 0.0f; rq[r] = 0.0f; }

#pragma unroll
    for (int e = 0; e < ETILES; ++e) {
      v8f ag = {}, av = {};
#pragma unroll
      for (int k = 0; k < KSLABS; ++k) ag = wmma_bf16(ah[k], load_bfrag(W4l, e, k, lane), ag);
#pragma unroll
      for (int k = 0; k < KSLABS; ++k) av = wmma_bf16(ah[k], load_bfrag(W5l, e, k, lane), av);
      const int col = e * 16 + ncol;
      float bg = b4[col], bv = b5[col];
      v8f y;
#pragma unroll
      for (int r = 0; r < 8; ++r) {
        float s  = 1.0f / (1.0f + __expf(-(ag[r] + bg)));   // sigmoid gate
        float v  = av[r] + bv;
        float xr = xt[(mb + r) * DDIM + col];               // f32 residual
        float yy = xr + s * v;
        y[r] = yy;
        rs[r] += yy;
        rq[r] += yy * yy;
      }
      yv[e] = y;
    }

    // Row-wise sums across the 16 lanes sharing each row (wave32, xor masks stay in group).
#pragma unroll
    for (int mask = 1; mask < 16; mask <<= 1) {
#pragma unroll
      for (int r = 0; r < 8; ++r) {
        rs[r] += __shfl_xor(rs[r], mask, 32);
        rq[r] += __shfl_xor(rq[r], mask, 32);
      }
    }
    float mean[8], rstd[8];
#pragma unroll
    for (int r = 0; r < 8; ++r) {
      float mu  = rs[r] * (1.0f / DDIM);
      float var = rq[r] * (1.0f / DDIM) - mu * mu;
      mean[r] = mu;
      rstd[r] = rsqrtf(var + 1e-5f);
    }

    float* op = out + (size_t)tok0 * DDIM;
#pragma unroll
    for (int e = 0; e < ETILES; ++e) {
      const int col = e * 16 + ncol;
      float gm = gamma[col], bt = beta[col];
#pragma unroll
      for (int r = 0; r < 8; ++r) {
        op[(mb + r) * DDIM + col] = (yv[e][r] - mean[r]) * rstd[r] * gm + bt;
      }
    }
  }
}

extern "C" void kernel_launch(void* const* d_in, const int* in_sizes, int n_in,
                              void* d_out, int out_size, void* d_ws, size_t ws_size,
                              hipStream_t stream) {
  (void)in_sizes; (void)n_in; (void)out_size; (void)d_ws; (void)ws_size;
  const float* x     = (const float*)d_in[0];
  const float* c     = (const float*)d_in[1];
  const float* W2    = (const float*)d_in[2];
  const float* b2    = (const float*)d_in[3];
  const float* W3    = (const float*)d_in[4];
  const float* W1    = (const float*)d_in[5];
  const float* b1    = (const float*)d_in[6];
  const float* W4    = (const float*)d_in[7];
  const float* b4    = (const float*)d_in[8];
  const float* W5    = (const float*)d_in[9];
  const float* b5    = (const float*)d_in[10];
  const float* gamma = (const float*)d_in[11];
  const float* beta  = (const float*)d_in[12];
  float* out = (float*)d_out;

  // Allow >64KB dynamic LDS (CDNA5 WGP has 320KB).
  (void)hipFuncSetAttribute((const void*)grn_fused,
                            hipFuncAttributeMaxDynamicSharedMemorySize,
                            (int)LDS_TOTAL);

  grn_fused<<<dim3(NGROUPS), dim3(WG_THREADS), LDS_TOTAL, stream>>>(
      x, c, W2, b2, W3, W1, b1, W4, b4, W5, b5, gamma, beta, out);
}
